// LightweightAttention_11364483465316
// MI455X (gfx1250) — matrix operand order repaired and checked by reading, not compiled, not run on hardware
//
#include <hip/hip_runtime.h>

// ---------------------------------------------------------------------------
// CDNA5 (gfx1250, wave32) fused lightweight-attention pipeline.
//   out = (context[B,D], attn_weights[B,H,T,T])  -- f32, concatenated flat.
// attended is only needed at t = T-1, so the full P@V einsum is skipped.
// Matmuls: v_wmma_f32_16x16x32_bf16.  B-operand of the QKV GEMM is staged
// through LDS with global_load_async_to_lds_b128 (ASYNCcnt) double-buffering.
// ---------------------------------------------------------------------------

typedef __attribute__((ext_vector_type(16))) __bf16 v16bf;
typedef __attribute__((ext_vector_type(8)))  __bf16 v8bf;
typedef __attribute__((ext_vector_type(8)))  float  v8f;

constexpr int   B_  = 16;
constexpr int   T_  = 1024;
constexpr int   D_  = 1024;
constexpr int   H_  = 8;
constexpr int   HD_ = 128;
constexpr int   BT_ = B_ * T_;
constexpr float NEGV  = -1.0e9f;
constexpr float SCALE = 0.08838834764831845f;   // 1/sqrt(HD)

// LDS B-tile: 64 n-rows x 32 k (bf16) = 64B payload/row, padded to 80B so the
// 16 fragment rows read in parallel map to distinct banks (stride 20 words).
constexpr int BROW = 80;

__device__ __forceinline__ unsigned short f32_to_bf16(float f) {
  unsigned u = __builtin_bit_cast(unsigned, f);
  unsigned r = u + 0x7FFFu + ((u >> 16) & 1u);    // round-to-nearest-even
  return (unsigned short)(r >> 16);
}
__device__ __forceinline__ float bf16u_to_f32(unsigned short h) {
  return __builtin_bit_cast(float, (unsigned)h << 16);
}

// Async global->LDS copy of 16B per lane (CDNA5, ASYNCcnt-tracked).
__device__ __forceinline__ void async_ld_b128(unsigned lds_off, const void* g) {
  unsigned long long ga = (unsigned long long)g;
  asm volatile("global_load_async_to_lds_b128 %0, %1, off"
               :: "v"(lds_off), "v"(ga) : "memory");
}
__device__ __forceinline__ void wait_async0() {
#if __has_builtin(__builtin_amdgcn_s_wait_asynccnt)
  __builtin_amdgcn_s_wait_asynccnt(0);
#else
  asm volatile("s_wait_asynccnt 0x0" ::: "memory");
#endif
}

// A fragment (contiguous k order: k = 16*half + s): 32 contiguous bytes per lane.
__device__ __forceinline__ v16bf load_a_frag(const __bf16* __restrict__ row,
                                             int kb, int half) {
  v8bf lo = *(const v8bf*)(row + kb + half * 16);
  v8bf hi = *(const v8bf*)(row + kb + half * 16 + 8);
  return __builtin_shufflevector(lo, hi,
      0, 1, 2, 3, 4, 5, 6, 7, 8, 9, 10, 11, 12, 13, 14, 15);
}
// B fragment from an LDS row: 32 contiguous bytes (16B-aligned halves).
__device__ __forceinline__ v16bf load_b_frag_lds(const char* base) {
  v8bf lo = *(const v8bf*)(base);
  v8bf hi = *(const v8bf*)(base + 16);
  return __builtin_shufflevector(lo, hi,
      0, 1, 2, 3, 4, 5, 6, 7, 8, 9, 10, 11, 12, 13, 14, 15);
}

// ---------------------------------------------------------------------------
// 1) elementwise f32 -> bf16
__global__ __launch_bounds__(256)
void cvt_bf16_kernel(const float* __restrict__ in, unsigned short* __restrict__ out,
                     size_t n) {
  size_t i = (size_t)blockIdx.x * blockDim.x + threadIdx.x;
  if (i < n) out[i] = f32_to_bf16(in[i]);
}

// 2) W[k][n] (DxD) -> WT[n][k] bf16, so B-fragments become contiguous loads.
__global__ __launch_bounds__(256)
void transpose_cvt_kernel(const float* __restrict__ W, unsigned short* __restrict__ WT) {
  size_t i = (size_t)blockIdx.x * blockDim.x + threadIdx.x;   // i = n*D + k
  int n = (int)(i / D_);
  int k = (int)(i % D_);
  WT[i] = f32_to_bf16(W[(size_t)k * D_ + n]);
}

// ---------------------------------------------------------------------------
// 3) Out[M=BT, N=D] = X[M,K=D] @ WT[N,K]^T  in bf16 with f32 accum.
//    Block: 8 waves x (32m x 64n) = 256m x 64n.  B tile (64n x 32k) is staged
//    in LDS once per block per K-step via async copies, double-buffered.
__global__ __launch_bounds__(256)
void qkv_gemm_bf16(const unsigned short* __restrict__ Xu,
                   const unsigned short* __restrict__ WTu,
                   unsigned short* __restrict__ Out) {
  __shared__ char Bsh[2][64 * BROW];          // 2 x 5120B

  const int tid  = threadIdx.x;
  const int lane = tid & 31;
  const int wave = tid >> 5;
  const int half = lane >> 4;                 // 0: lanes 0-15, 1: lanes 16-31
  const int lm   = lane & 15;
  const int m0 = blockIdx.y * 256 + wave * 32;
  const int n0 = blockIdx.x * 64;

  const __bf16* X  = (const __bf16*)Xu;
  const __bf16* WT = (const __bf16*)WTu;

  // Copy role: 256 threads = 64 rows x 4 chunks of 16B.
  const int crow = tid >> 2;                  // 0..63
  const int cchk = tid & 3;                   // 0..3
  const __bf16* gsrc = WT + (size_t)(n0 + crow) * D_ + cchk * 8;
  const unsigned ldsoff0 = (unsigned)(unsigned long long)&Bsh[0][crow * BROW + cchk * 16];
  const unsigned ldsoff1 = (unsigned)(unsigned long long)&Bsh[1][crow * BROW + cchk * 16];

  // Prologue: stage first B tile.
  async_ld_b128(ldsoff0, gsrc);

  v8f acc[8] = {v8f{}, v8f{}, v8f{}, v8f{}, v8f{}, v8f{}, v8f{}, v8f{}};
  const __bf16* arow0 = X + (size_t)(m0 + lm) * D_;
  const __bf16* arow1 = X + (size_t)(m0 + 16 + lm) * D_;

  wait_async0();
  __syncthreads();

  int buf = 0;
  for (int kb = 0; kb < D_; kb += 32, buf ^= 1) {
    if (kb + 32 < D_)                          // stage next tile (other buffer)
      async_ld_b128(buf ? ldsoff0 : ldsoff1, gsrc + kb + 32);

    __builtin_prefetch((const void*)(arow0 + kb + 256), 0, 3);
    __builtin_prefetch((const void*)(arow1 + kb + 256), 0, 3);

    v16bf a0 = load_a_frag(arow0, kb, half);
    v16bf a1 = load_a_frag(arow1, kb, half);

    v16bf bfrag[4];
#pragma unroll
    for (int t = 0; t < 4; ++t)
      bfrag[t] = load_b_frag_lds(&Bsh[buf][(t * 16 + lm) * BROW + half * 32]);

#pragma unroll
    for (int t = 0; t < 4; ++t)
      acc[t] = __builtin_amdgcn_wmma_f32_16x16x32_bf16(
          false, a0, false, bfrag[t], (short)0, acc[t], false, false);
#pragma unroll
    for (int t = 0; t < 4; ++t)
      acc[4 + t] = __builtin_amdgcn_wmma_f32_16x16x32_bf16(
          false, a1, false, bfrag[t], (short)0, acc[4 + t], false, false);

    wait_async0();                             // own async copies done
    __syncthreads();                           // whole tile visible / reads done
  }

  // C layout: lane = col n0+t*16+lm, rows (g*16 + half*8 + r).
#pragma unroll
  for (int g = 0; g < 2; ++g) {
#pragma unroll
    for (int t = 0; t < 4; ++t) {
#pragma unroll
      for (int r = 0; r < 8; ++r) {
        int m = m0 + g * 16 + half * 8 + r;
        int n = n0 + t * 16 + lm;
        Out[(size_t)m * D_ + n] = f32_to_bf16(acc[g * 4 + t][r]);
      }
    }
  }
}

// ---------------------------------------------------------------------------
// 4) Per (b,h,q-tile): S = Q Kt / sqrt(HD) with causal mask, row softmax,
//    write P (f32) to attn output.  8 waves share 16 rows x 1024 keys in LDS.
__global__ __launch_bounds__(256)
void attn_softmax(const unsigned short* __restrict__ Qu,
                  const unsigned short* __restrict__ Ku,
                  float* __restrict__ attn) {
  __shared__ float S[16][T_ + 8];

  const int lane = threadIdx.x & 31;
  const int wave = threadIdx.x >> 5;
  const int half = lane >> 4;
  const int lm   = lane & 15;
  const int qt = blockIdx.x, h = blockIdx.y, b = blockIdx.z;
  const int q0 = qt * 16;
  const size_t headoff = (size_t)b * T_ * D_ + (size_t)h * HD_;

  const __bf16* Q  = (const __bf16*)Qu;
  const __bf16* Kc = (const __bf16*)Ku;

  // Preload the 4 HD-dim A-fragments of the Q tile; reused by all key tiles.
  v16bf aq[4];
  const __bf16* qrow = Q + headoff + (size_t)(q0 + lm) * D_;
#pragma unroll
  for (int db = 0; db < 4; ++db)
    aq[db] = load_a_frag(qrow, db * 32, half);

  for (int kt = wave; kt < T_ / 16; kt += 8) {
    const int k0 = kt * 16;
    v8f s = {};
    if (k0 <= q0 + 15) {   // wave-uniform: EXEC stays all-ones around WMMA
      const __bf16* krow = Kc + headoff + (size_t)(k0 + lm) * D_ + half * 16;
      v16bf bk[4];
#pragma unroll
      for (int db = 0; db < 4; ++db)
        bk[db] = *(const v16bf*)(krow + db * 32);
#pragma unroll
      for (int db = 0; db < 4; ++db)
        s = __builtin_amdgcn_wmma_f32_16x16x32_bf16(
            false, aq[db], false, bk[db], (short)0, s, false, false);
    }
#pragma unroll
    for (int r = 0; r < 8; ++r) {
      int row = half * 8 + r;
      int k   = k0 + lm;
      S[row][k] = (k <= q0 + row) ? s[r] * SCALE : NEGV;
    }
  }
  __syncthreads();

  // Softmax: each wave handles 2 of the 16 rows; lanes stride the 1024 keys.
#pragma unroll
  for (int rr = 0; rr < 2; ++rr) {
    const int row = wave * 2 + rr;
    const int q   = q0 + row;
    float mx = -3.4e38f;
    for (int k = lane; k < T_; k += 32) mx = fmaxf(mx, S[row][k]);
#pragma unroll
    for (int off = 16; off > 0; off >>= 1) mx = fmaxf(mx, __shfl_xor(mx, off, 32));
    float sum = 0.f;
    for (int k = lane; k < T_; k += 32) {
      float e = __expf(S[row][k] - mx);
      S[row][k] = e;
      sum += e;
    }
#pragma unroll
    for (int off = 16; off > 0; off >>= 1) sum += __shfl_xor(sum, off, 32);
    const float inv = 1.0f / sum;
    float* orow = attn + ((size_t)((b * H_ + h) * T_ + q)) * T_;
    for (int k = lane; k < T_; k += 32) orow[k] = S[row][k] * inv;
  }
}

// ---------------------------------------------------------------------------
// 5) context[b, h*HD+d] = sum_k P[b,h,T-1,k] * V[b,k,h*HD+d]   (tiny)
__global__ __launch_bounds__(128)
void context_out(const float* __restrict__ attn,
                 const unsigned short* __restrict__ Vu,
                 float* __restrict__ ctx) {
  const int d = threadIdx.x;          // 0..127
  const int h = blockIdx.x, b = blockIdx.y;
  const float* p = attn + ((size_t)((b * H_ + h) * T_ + (T_ - 1))) * T_;
  const unsigned short* vcol = Vu + (size_t)b * T_ * D_ + (size_t)h * HD_ + d;
  float acc = 0.f;
  for (int k = 0; k < T_; ++k)
    acc += p[k] * bf16u_to_f32(vcol[(size_t)k * D_]);
  ctx[(size_t)b * D_ + h * HD_ + d] = acc;
}

// ---------------------------------------------------------------------------
extern "C" void kernel_launch(void* const* d_in, const int* in_sizes, int n_in,
                              void* d_out, int out_size, void* d_ws, size_t ws_size,
                              hipStream_t stream) {
  const float* x  = (const float*)d_in[0];
  const float* Wq = (const float*)d_in[1];
  const float* Wk = (const float*)d_in[2];
  const float* Wv = (const float*)d_in[3];

  const size_t NXB = (size_t)BT_ * D_;   // 16,777,216 elements
  const size_t NW  = (size_t)D_ * D_;    //  1,048,576 elements

  // Workspace layout (bf16 as u16): x | WqT | WkT | WvT | Q | K | V  (~140 MB)
  unsigned short* ws  = (unsigned short*)d_ws;
  unsigned short* xb  = ws;
  unsigned short* wqT = xb  + NXB;
  unsigned short* wkT = wqT + NW;
  unsigned short* wvT = wkT + NW;
  unsigned short* Qb  = wvT + NW;
  unsigned short* Kb  = Qb  + NXB;
  unsigned short* Vb  = Kb  + NXB;

  float* out  = (float*)d_out;
  float* ctx  = out;                         // B*D floats
  float* attn = out + (size_t)B_ * D_;       // B*H*T*T floats

  cvt_bf16_kernel<<<dim3((unsigned)(NXB / 256)), 256, 0, stream>>>(x, xb, NXB);
  transpose_cvt_kernel<<<dim3((unsigned)(NW / 256)), 256, 0, stream>>>(Wq, wqT);
  transpose_cvt_kernel<<<dim3((unsigned)(NW / 256)), 256, 0, stream>>>(Wk, wkT);
  transpose_cvt_kernel<<<dim3((unsigned)(NW / 256)), 256, 0, stream>>>(Wv, wvT);

  dim3 ggrid(D_ / 64, BT_ / 256);            // (16, 64)
  qkv_gemm_bf16<<<ggrid, 256, 0, stream>>>(xb, wqT, Qb);
  qkv_gemm_bf16<<<ggrid, 256, 0, stream>>>(xb, wkT, Kb);
  qkv_gemm_bf16<<<ggrid, 256, 0, stream>>>(xb, wvT, Vb);

  attn_softmax<<<dim3(T_ / 16, H_, B_), 256, 0, stream>>>(Qb, Kb, attn);
  context_out<<<dim3(H_, B_), HD_, 0, stream>>>(attn, Vb, ctx);
}
